// FeaturePropagation_36352603193824
// MI455X (gfx1250) — matrix operand order repaired and checked by reading, not compiled, not run on hardware
//
#include <hip/hip_runtime.h>
#include <hip/hip_fp16.h>
#include <stdint.h>

typedef __attribute__((ext_vector_type(16))) _Float16 v16h;
typedef __attribute__((ext_vector_type(8)))  _Float16 v8h;
typedef __attribute__((ext_vector_type(8)))  float    v8f;

#define B_    16
#define N1_   4096
#define N2_   1024
#define C1_   128
#define C2_   256
#define H_    256
#define CIN_  384
#define M_    (B_ * N1_)   // 65536 rows

// ---------------------------------------------------------------------------
// gfx1250 async-copy helpers (ASYNCcnt-tracked global->LDS DMA)
// ---------------------------------------------------------------------------
__device__ __forceinline__ void async_load_b128_to_lds(unsigned lds_byte_off,
                                                       const void* gaddr) {
    const unsigned long long ga = (unsigned long long)(uintptr_t)gaddr;
    asm volatile("global_load_async_to_lds_b128 %0, %1, off"
                 :: "v"(lds_byte_off), "v"(ga) : "memory");
}
__device__ __forceinline__ void wait_asynccnt0() {
#if __has_builtin(__builtin_amdgcn_s_wait_asynccnt)
    __builtin_amdgcn_s_wait_asynccnt(0);
#else
    asm volatile("s_wait_asynccnt 0" ::: "memory");
#endif
}

// ---------------------------------------------------------------------------
// K1: convert W1 [256x384] and W2 [256x256] fp32 -> f16, row-major
// ---------------------------------------------------------------------------
__global__ void cvt_weights_kernel(const float* __restrict__ W1,
                                   const float* __restrict__ W2,
                                   _Float16* __restrict__ W1h,
                                   _Float16* __restrict__ W2h) {
    int i = blockIdx.x * 256 + threadIdx.x;
    if (i < H_ * CIN_) W1h[i] = (_Float16)W1[i];
    if (i < H_ * H_)   W2h[i] = (_Float16)W2[i];
}

// ---------------------------------------------------------------------------
// K2: 3-NN inverse-distance interpolation + concat -> featsH [M, 384] f16
// ---------------------------------------------------------------------------
__global__ void knn_interp_concat_kernel(const float* __restrict__ c1,
                                         const float* __restrict__ c2,
                                         const float* __restrict__ f1,
                                         const float* __restrict__ f2,
                                         _Float16* __restrict__ featsH) {
    __shared__ float q[3][N2_];                 // 12 KB of the 320 KB WGP LDS
    const int b     = blockIdx.x / (N1_ / 256);
    const int chunk = blockIdx.x % (N1_ / 256);
    const int t     = threadIdx.x;

    const float* c2b = c2 + (size_t)b * 3 * N2_;
    for (int i = t; i < 3 * N2_; i += 256) q[i / N2_][i % N2_] = c2b[i];
    __syncthreads();

    const int n = chunk * 256 + t;
    const float px = c1[((size_t)b * 3 + 0) * N1_ + n];
    const float py = c1[((size_t)b * 3 + 1) * N1_ + n];
    const float pz = c1[((size_t)b * 3 + 2) * N1_ + n];

    float d0 = 3.4e38f, d1 = 3.4e38f, d2 = 3.4e38f;
    int   i0 = 0, i1 = 0, i2 = 0;
    for (int j = 0; j < N2_; ++j) {
        const float dx = px - q[0][j];
        const float dy = py - q[1][j];
        const float dz = pz - q[2][j];
        const float d  = dx * dx + dy * dy + dz * dz;
        if (d < d0)      { d2 = d1; i2 = i1; d1 = d0; i1 = i0; d0 = d; i0 = j; }
        else if (d < d1) { d2 = d1; i2 = i1; d1 = d;  i1 = j; }
        else if (d < d2) { d2 = d;  i2 = j; }
    }
    const float r0 = 1.0f / (d0 + 1e-8f);
    const float r1 = 1.0f / (d1 + 1e-8f);
    const float r2 = 1.0f / (d2 + 1e-8f);
    const float rs = 1.0f / (r0 + r1 + r2);
    const float w0 = r0 * rs, w1 = r1 * rs, w2 = r2 * rs;

    _Float16* row = featsH + (size_t)(b * N1_ + n) * CIN_;
    const float* f1b = f1 + (size_t)b * C1_ * N1_ + n;
    #pragma unroll 4
    for (int c = 0; c < C1_; ++c)
        row[c] = (_Float16)f1b[(size_t)c * N1_];

    const float* f2b = f2 + (size_t)b * C2_ * N2_;
    #pragma unroll 4
    for (int c = 0; c < C2_; ++c) {
        const float* fc = f2b + (size_t)c * N2_;
        const float v = w0 * fc[i0] + w1 * fc[i1] + w2 * fc[i2];
        row[C1_ + c] = (_Float16)v;
    }
}

// ---------------------------------------------------------------------------
// GEMM via v_wmma_f32_16x16x32_f16.
//   Y[m,n] = sum_k A[m,k] * W[n,k]   (A row-major [M,K] f16, W row-major [256,K] f16)
//
// Block = 256 threads (8 waves). Block tile: 256(M) x 64(N).
//   - The 64xK weight slice is staged once per block into LDS with
//     global_load_async_to_lds_b128 (ASYNCcnt), rows padded +16 halves to
//     spread LDS banks; all 8 waves read B fragments from LDS.
//   - Each wave computes a 32(M) x 64(N) tile: 8 accumulators, K-step 32,
//     A fragments streamed from global (pipelined b128 loads).
// Fragment layouts per CDNA5 ISA 7.12.2 (16-bit A 16x32 / B 32x16, f32 D).
// ---------------------------------------------------------------------------
template <int K>
__global__ void gemm_wmma_kernel(const _Float16* __restrict__ A,
                                 const _Float16* __restrict__ Wg,
                                 float* __restrict__ Y) {
    constexpr int LSTR = K + 16;                 // padded LDS row stride (halves)
    __shared__ _Float16 wlds[64 * LSTR];

    const int wave = threadIdx.x >> 5;           // 8 waves / block
    const int lane = threadIdx.x & 31;
    const int lo   = lane & 15;
    const int hi   = lane >> 4;
    const int mbase = blockIdx.x * 256 + wave * 32;
    const int nbase = blockIdx.y * 64;

    // ---- async-stage the 64 x K weight slice into LDS ----
    constexpr int CH = K / 8;                    // b128 chunks per row
    for (int i = threadIdx.x; i < 64 * CH; i += 256) {
        const int row = i / CH, c8 = i % CH;
        const _Float16* gp = Wg + (size_t)(nbase + row) * K + c8 * 8;
        const unsigned ldsoff =
            (unsigned)(uintptr_t)&wlds[row * LSTR + c8 * 8];
        async_load_b128_to_lds(ldsoff, gp);
    }
    wait_asynccnt0();
    __syncthreads();

    v8f acc[8] = {v8f{}, v8f{}, v8f{}, v8f{}, v8f{}, v8f{}, v8f{}, v8f{}};
    const _Float16* arow0 = A + (size_t)(mbase + lo) * K;
    const _Float16* arow1 = A + (size_t)(mbase + 16 + lo) * K;

    for (int k0 = 0; k0 < K; k0 += 32) {
        const v8h a0lo = *(const v8h*)(arow0 + k0 + hi * 8);
        const v8h a0hi = *(const v8h*)(arow0 + k0 + 16 + hi * 8);
        const v8h a1lo = *(const v8h*)(arow1 + k0 + hi * 8);
        const v8h a1hi = *(const v8h*)(arow1 + k0 + 16 + hi * 8);
        v16h a0, a1;
        #pragma unroll
        for (int i = 0; i < 8; ++i) {
            a0[i] = a0lo[i]; a0[i + 8] = a0hi[i];
            a1[i] = a1lo[i]; a1[i + 8] = a1hi[i];
        }

        #pragma unroll
        for (int t = 0; t < 4; ++t) {
            const _Float16* wrow = wlds + (size_t)(t * 16 + lo) * LSTR + k0 + hi * 8;
            const v8h b0 = *(const v8h*)(wrow);
            const v8h b1 = *(const v8h*)(wrow + 16);
            v16h bb;
            #pragma unroll
            for (int i = 0; i < 8; ++i) { bb[i] = b0[i]; bb[i + 8] = b1[i]; }
            acc[t] = __builtin_amdgcn_wmma_f32_16x16x32_f16(
                false, a0, false, bb, (short)0, acc[t], false, false);
            acc[4 + t] = __builtin_amdgcn_wmma_f32_16x16x32_f16(
                false, a1, false, bb, (short)0, acc[4 + t], false, false);
        }
    }

    #pragma unroll
    for (int s = 0; s < 2; ++s) {
        #pragma unroll
        for (int t = 0; t < 4; ++t) {
            #pragma unroll
            for (int r = 0; r < 8; ++r) {
                Y[(size_t)(mbase + s * 16 + hi * 8 + r) * H_ +
                  (nbase + t * 16 + lo)] = acc[s * 4 + t][r];
            }
        }
    }
}

// ---------------------------------------------------------------------------
// deterministic per-channel partial sums (256 blocks x 256 rows each)
// ---------------------------------------------------------------------------
__global__ void bn_partial_kernel(const float* __restrict__ Y,
                                  float* __restrict__ psum,
                                  float* __restrict__ psq) {
    const int t   = threadIdx.x;          // channel
    const int blk = blockIdx.x;           // row chunk
    const size_t row0 = (size_t)blk * 256;
    float s = 0.f, s2 = 0.f;
    for (int r = 0; r < 256; ++r) {
        const float v = Y[(row0 + r) * H_ + t];
        s += v; s2 += v * v;
    }
    psum[blk * H_ + t] = s;
    psq[blk * H_ + t]  = s2;
}

// ---------------------------------------------------------------------------
// finalize BN params (fixed-order reduction -> deterministic)
// ---------------------------------------------------------------------------
__global__ void bn_finalize_kernel(const float* __restrict__ psum,
                                   const float* __restrict__ psq,
                                   const float* __restrict__ gamma,
                                   const float* __restrict__ beta,
                                   float* __restrict__ scale,
                                   float* __restrict__ shift) {
    const int c = threadIdx.x;
    float s = 0.f, s2 = 0.f;
    for (int i = 0; i < 256; ++i) { s += psum[i * H_ + c]; s2 += psq[i * H_ + c]; }
    const float inv  = 1.0f / (float)M_;
    const float mean = s * inv;
    const float var  = s2 * inv - mean * mean;
    const float sc   = gamma[c] * rsqrtf(var + 1e-5f);
    scale[c] = sc;
    shift[c] = beta[c] - mean * sc;
}

// ---------------------------------------------------------------------------
// BN + ReLU -> f16 activations for GEMM2 (c == threadIdx.x since H_==256)
// ---------------------------------------------------------------------------
__global__ void bn_apply_h_kernel(const float* __restrict__ Y,
                                  const float* __restrict__ scale,
                                  const float* __restrict__ shift,
                                  _Float16* __restrict__ Xh) {
    const size_t i = (size_t)blockIdx.x * 256 + threadIdx.x;
    const int c = threadIdx.x;
    float v = Y[i] * scale[c] + shift[c];
    Xh[i] = (_Float16)(v > 0.f ? v : 0.f);
}

// ---------------------------------------------------------------------------
// BN + ReLU + transpose [M,H] -> out [B, H, N1]
// ---------------------------------------------------------------------------
__global__ void bn_out_kernel(const float* __restrict__ Y,
                              const float* __restrict__ scale,
                              const float* __restrict__ shift,
                              float* __restrict__ out) {
    const int nchunk = blockIdx.x & 15;
    const int c      = (blockIdx.x >> 4) & 255;
    const int b      = blockIdx.x >> 12;
    const int n      = nchunk * 256 + threadIdx.x;
    float v = Y[(size_t)(b * N1_ + n) * H_ + c];
    v = v * scale[c] + shift[c];
    out[((size_t)b * H_ + c) * N1_ + n] = fmaxf(v, 0.f);
}

// ---------------------------------------------------------------------------
extern "C" void kernel_launch(void* const* d_in, const int* in_sizes, int n_in,
                              void* d_out, int out_size, void* d_ws, size_t ws_size,
                              hipStream_t stream) {
    const float* c1  = (const float*)d_in[0];   // [B,3,N1]
    const float* c2  = (const float*)d_in[1];   // [B,3,N2]
    const float* f1  = (const float*)d_in[2];   // [B,C1,N1]
    const float* f2  = (const float*)d_in[3];   // [B,C2,N2]
    const float* W1  = (const float*)d_in[4];   // [H,CIN]
    // d_in[5] = b1 : conv bias cancels exactly through train-mode BN -> skipped
    const float* g1  = (const float*)d_in[6];
    const float* be1 = (const float*)d_in[7];
    const float* W2  = (const float*)d_in[8];   // [H,H]
    // d_in[9] = b2 : skipped (cancels in BN)
    const float* g2  = (const float*)d_in[10];
    const float* be2 = (const float*)d_in[11];
    float* out = (float*)d_out;                 // [B,H,N1]

    // workspace layout (all 256B aligned)
    char* ws = (char*)d_ws;
    _Float16* featsH = (_Float16*)ws;                                   // 48 MB
    float*    Yb     = (float*)(ws + (size_t)M_ * CIN_ * 2);            // 64 MB (y1, reused as y2)
    _Float16* Xh     = (_Float16*)((char*)Yb + (size_t)M_ * H_ * 4);    // 32 MB
    char* aux        = (char*)Xh + (size_t)M_ * H_ * 2;
    _Float16* W1h = (_Float16*)aux;                    aux += (size_t)H_ * CIN_ * 2;
    _Float16* W2h = (_Float16*)aux;                    aux += (size_t)H_ * H_ * 2;
    float* psum   = (float*)aux;                       aux += 256 * H_ * 4;
    float* psq    = (float*)aux;                       aux += 256 * H_ * 4;
    float* scale1 = (float*)aux;                       aux += H_ * 4;
    float* shift1 = (float*)aux;                       aux += H_ * 4;
    float* scale2 = (float*)aux;                       aux += H_ * 4;
    float* shift2 = (float*)aux;

    // 1. weights -> f16
    cvt_weights_kernel<<<(H_ * CIN_ + 255) / 256, 256, 0, stream>>>(W1, W2, W1h, W2h);
    // 2. knn + interp + concat -> featsH [M, 384]
    knn_interp_concat_kernel<<<B_ * (N1_ / 256), 256, 0, stream>>>(c1, c2, f1, f2, featsH);
    // 3. GEMM1: [M,384] x [384,256] -> Yb
    gemm_wmma_kernel<CIN_><<<dim3(M_ / 256, 4), 256, 0, stream>>>(featsH, W1h, Yb);
    // 4-5. BN1 stats
    bn_partial_kernel<<<256, 256, 0, stream>>>(Yb, psum, psq);
    bn_finalize_kernel<<<1, 256, 0, stream>>>(psum, psq, g1, be1, scale1, shift1);
    // 6. BN1 + ReLU -> f16
    bn_apply_h_kernel<<<M_, 256, 0, stream>>>(Yb, scale1, shift1, Xh);
    // 7. GEMM2: [M,256] x [256,256] -> Yb (y1 dead)
    gemm_wmma_kernel<H_><<<dim3(M_ / 256, 4), 256, 0, stream>>>(Xh, W2h, Yb);
    // 8-9. BN2 stats
    bn_partial_kernel<<<256, 256, 0, stream>>>(Yb, psum, psq);
    bn_finalize_kernel<<<1, 256, 0, stream>>>(psum, psq, g2, be2, scale2, shift2);
    // 10. BN2 + ReLU + transpose -> out [B,256,N1]
    bn_out_kernel<<<B_ * H_ * (N1_ / 256), 256, 0, stream>>>(Yb, scale2, shift2, out);
}
